// SimpleGatedElman_738734375262
// MI455X (gfx1250) — matrix operand order repaired and checked by reading, not compiled, not run on hardware
//
#include <hip/hip_runtime.h>
#include <hip/hip_bf16.h>
#include <cstdint>

// ---------------- problem sizes (fixed by the reference) ----------------
#define D_    1024
#define T_    1024
#define B_    4
#define L_    4
#define V_    32000
#define BT_   4096            // B_*T_
#define EPS_  1e-5f

// persistent scan configuration
#define GS_   32              // scan workgroups (each owns D_/GS_ output columns)
#define NSL_  (D_/GS_)        // 32 columns per scan WG
#define WHPAD 8               // pad so LDS row stride is 16B-aligned, 2-way banked
#define WHROW (D_ + WHPAD)    // 1032

typedef __attribute__((ext_vector_type(16))) _Float16 v16h;
typedef __attribute__((ext_vector_type(8)))  float    v8f;

// low 32 bits of a flat pointer into __shared__ == LDS byte address
// (LDS aperture keeps the offset in addr[31:0], ISA 10.2)
__device__ __forceinline__ unsigned lds_lo32(const void* p) {
  return (unsigned)(uintptr_t)p;
}

// CDNA5 async DMA: 16B per lane, global -> LDS, tracked by ASYNCcnt
__device__ __forceinline__ void async_b128(unsigned lds_addr, const void* gaddr) {
  asm volatile("global_load_async_to_lds_b128 %0, %1, off"
               :: "v"(lds_addr), "v"(gaddr) : "memory");
}

// ---------------------------------------------------------------------------
// f32 -> f16 elementwise convert (all sizes here are multiples of 4)
// ---------------------------------------------------------------------------
__global__ __launch_bounds__(256) void k_cvt_f16(const float* __restrict__ in,
                                                 _Float16* __restrict__ out, int n) {
  int i = (blockIdx.x * 256 + threadIdx.x) * 4;
  if (i + 3 < n) {
    float4 v = *(const float4*)(in + i);
    out[i + 0] = (_Float16)v.x;
    out[i + 1] = (_Float16)v.y;
    out[i + 2] = (_Float16)v.z;
    out[i + 3] = (_Float16)v.w;
  }
}

// ---------------------------------------------------------------------------
// embedding gather: h[bt, :] = embed[x[bt], :]
// ---------------------------------------------------------------------------
__global__ __launch_bounds__(256) void k_embed_gather(const int* __restrict__ x,
                                                      const float* __restrict__ embed,
                                                      float* __restrict__ h) {
  const int bt = blockIdx.x;
  const int idx = x[bt];
  const float4* src = (const float4*)(embed + (size_t)idx * D_);
  float4* dst = (float4*)(h + (size_t)bt * D_);
  for (int i = threadIdx.x; i < D_ / 4; i += 256) dst[i] = src[i];
}

// ---------------------------------------------------------------------------
// LayerNorm over D_=1024, f32 in, f16 out (feeds WMMA GEMMs). 1 block/token.
// ---------------------------------------------------------------------------
__global__ __launch_bounds__(256) void k_layernorm_f16(const float* __restrict__ x,
                                                       const float* __restrict__ sc,
                                                       const float* __restrict__ bi,
                                                       _Float16* __restrict__ out) {
  const int row = blockIdx.x;
  const float* xr = x + (size_t)row * D_;
  float s = 0.f, s2 = 0.f;
  for (int i = threadIdx.x; i < D_; i += 256) {
    float v = xr[i];
    s += v;
    s2 += v * v;
  }
  // wave32 reduction then cross-wave via LDS
  for (int off = 16; off > 0; off >>= 1) {
    s  += __shfl_down(s, off, 32);
    s2 += __shfl_down(s2, off, 32);
  }
  __shared__ float rs[8], rs2[8];
  const int wv = threadIdx.x >> 5, ln = threadIdx.x & 31;
  if (ln == 0) { rs[wv] = s; rs2[wv] = s2; }
  __syncthreads();
  if (threadIdx.x == 0) {
    float a = 0.f, b = 0.f;
    for (int i = 0; i < 8; ++i) { a += rs[i]; b += rs2[i]; }
    rs[0] = a; rs2[0] = b;
  }
  __syncthreads();
  const float mean = rs[0] * (1.f / D_);
  const float var  = rs2[0] * (1.f / D_) - mean * mean;
  const float rstd = rsqrtf(var + EPS_);
  _Float16* orow = out + (size_t)row * D_;
  for (int i = threadIdx.x; i < D_; i += 256)
    orow[i] = (_Float16)((xr[i] - mean) * rstd * sc[i] + bi[i]);
}

// ---------------------------------------------------------------------------
// WMMA GEMM:  C[M,N] (f32) = A[M,K] (f16, row-major) @ W[N,K]^T (f16, row-major)
// Block tile 128x128, 256 threads = 8 waves (4 M-waves x 2 N-waves),
// wave tile 32x64 = 2x4 fragments of v_wmma_f32_16x16x32_f16, K step 32.
// Data movement: double-buffered GLOBAL_LOAD_ASYNC_TO_LDS_B128 (ASYNCcnt),
// next tile's DMA overlaps current tile's WMMAs; global_prefetch two ahead.
// M,N,K are multiples of 128/128/32 by construction.
// ---------------------------------------------------------------------------
__global__ __launch_bounds__(256) void k_gemm_wmma(const _Float16* __restrict__ A,
                                                   const _Float16* __restrict__ W,
                                                   float* __restrict__ C,
                                                   int M, int N, int K) {
  __shared__ _Float16 lA[2][128][40];  // 32 cols + 8 pad (conflict-free b128 reads)
  __shared__ _Float16 lB[2][128][40];  // lB[b][n][k] = W[n0+n][k0+k]

  const int m0 = blockIdx.y * 128;
  const int n0 = blockIdx.x * 128;
  const int tid  = threadIdx.x;
  const int lane = tid & 31;
  const int wave = tid >> 5;
  const int wm = (wave & 3) * 32;   // wave row offset inside block tile
  const int wn = (wave >> 2) * 64;  // wave col offset inside block tile
  const int lrow  = tid >> 1;       // 0..127: staging row
  const int lhalf = (tid & 1) * 16; // 0 or 16: staging K half
  const int hi    = lane >> 4;      // 0: lanes 0-15, 1: lanes 16-31
  const int lm    = lane & 15;

  // per-thread staging addresses (two 16B beats per matrix per tile)
  const _Float16* gA = A + (size_t)(m0 + lrow) * K + lhalf;
  const _Float16* gB = W + (size_t)(n0 + lrow) * K + lhalf;
  const unsigned laOff[2] = { lds_lo32(&lA[0][lrow][lhalf]), lds_lo32(&lA[1][lrow][lhalf]) };
  const unsigned lbOff[2] = { lds_lo32(&lB[0][lrow][lhalf]), lds_lo32(&lB[1][lrow][lhalf]) };

  v8f acc[2][4] = {};
  union HV { uint4 q[2]; v16h v; };

  const int nt = K / 32;

  // prologue: DMA tile 0 into buffer 0
  async_b128(laOff[0],      gA);
  async_b128(laOff[0] + 16, gA + 8);
  async_b128(lbOff[0],      gB);
  async_b128(lbOff[0] + 16, gB + 8);

  for (int it = 0; it < nt; ++it) {
    const int buf = it & 1;
    if (it + 1 < nt) {
      // kick DMA for the next tile into the other buffer
      const int nb = buf ^ 1;
      const _Float16* a1 = gA + (size_t)(it + 1) * 32;
      const _Float16* b1 = gB + (size_t)(it + 1) * 32;
      async_b128(laOff[nb],      a1);
      async_b128(laOff[nb] + 16, a1 + 8);
      async_b128(lbOff[nb],      b1);
      async_b128(lbOff[nb] + 16, b1 + 8);
      if (it + 2 < nt) {  // GL2 prefetch two tiles ahead
        __builtin_prefetch(gA + (size_t)(it + 2) * 32, 0, 1);
        __builtin_prefetch(gB + (size_t)(it + 2) * 32, 0, 1);
      }
      // current tile complete when only the 4 just-issued ops remain (in-order)
      asm volatile("s_wait_asynccnt 0x4" ::: "memory");
    } else {
      asm volatile("s_wait_asynccnt 0x0" ::: "memory");
    }
    __syncthreads();

    // Build fragments per ISA 7.12.2 layouts.
    // A (16x32 f16): lane = hi*16+m, half h -> k = (h&7) + hi*8 + (h>=8)*16
    v16h af[2];
#pragma unroll
    for (int i = 0; i < 2; ++i) {
      HV u;
      const _Float16* src = &lA[buf][wm + i * 16 + lm][0];
      u.q[0] = *(const uint4*)(src + hi * 8);        // h=0..7  -> k = hi*8 + 0..7
      u.q[1] = *(const uint4*)(src + 16 + hi * 8);   // h=8..15 -> k = 16 + hi*8 + 0..7
      af[i] = u.v;
    }
    // B (32x16 f16): lane = hi*16+n, half h -> k = (h&7) + hi*8 + (h>=8)*16
    v16h bf[4];
#pragma unroll
    for (int j = 0; j < 4; ++j) {
      HV u;
      const _Float16* src = &lB[buf][wn + j * 16 + lm][0];
      u.q[0] = *(const uint4*)(src + hi * 8);        // k = hi*8 + 0..7
      u.q[1] = *(const uint4*)(src + 16 + hi * 8);   // k = 16 + hi*8 + 0..7
      bf[j] = u.v;
    }

#pragma unroll
    for (int i = 0; i < 2; ++i)
#pragma unroll
      for (int j = 0; j < 4; ++j)
        acc[i][j] = __builtin_amdgcn_wmma_f32_16x16x32_f16(
            false, af[i], false, bf[j], (short)0, acc[i][j], false, false);

    __syncthreads();  // all waves done with buf before it is DMA'd again (it+2)
  }

  // C/D layout: VGPR r holds C[m0+wm+i*16 + hi*8 + r][n0+wn+j*16+lm]
#pragma unroll
  for (int i = 0; i < 2; ++i)
#pragma unroll
    for (int j = 0; j < 4; ++j) {
      const int n = n0 + wn + j * 16 + lm;
      const int mb = m0 + wm + i * 16 + hi * 8;
#pragma unroll
      for (int r = 0; r < 8; ++r)
        C[(size_t)(mb + r) * N + n] = acc[i][j][r];
    }
}

// ---------------------------------------------------------------------------
// Persistent gated-Elman scan. GS_=32 cooperative workgroups, 128 threads each.
// Each WG owns NSL_=32 output columns and keeps its Wh slice (f16) in LDS.
// Per step: pre = wx_t + hprev @ Wh^T (slice), hnew = tanh(pre),
//           out = sigm(z_h)*hnew + sigm(z_x)*wx_t, h += out (residual),
// h-state is double-buffered in global; steps separated by an agent-scope
// split barrier (release fence -> atomic arrive -> acquire spin + s_sleep).
// ---------------------------------------------------------------------------
__global__ __launch_bounds__(128) void k_scan(const float* __restrict__ wx,
                                              const float* __restrict__ z,
                                              const _Float16* __restrict__ Wh,
                                              float* __restrict__ h,
                                              float* __restrict__ hstate,  // [2][B_*D_]
                                              int* __restrict__ sync_cnt,
                                              int sync_base) {
  __shared__ _Float16 whs[NSL_][WHROW];  // ~66 KB of the WGP's 320 KB LDS
  const int tid = threadIdx.x;
  const int n0 = blockIdx.x * NSL_;

  // preload Wh slice (rows n0..n0+31), 16B vectors
  for (int i = tid; i < NSL_ * (D_ / 8); i += 128) {
    const int r = i / (D_ / 8);
    const int c = (i % (D_ / 8)) * 8;
    *(uint4*)&whs[r][c] = *(const uint4*)&Wh[(size_t)(n0 + r) * D_ + c];
  }
  __syncthreads();

  const int nl = tid & (NSL_ - 1);  // 0..31
  const int b  = tid >> 5;          // 0..3
  const int n  = n0 + nl;
  const _Float16* wr = &whs[nl][0];

  for (int t = 0; t < T_; ++t) {
    const size_t bt = (size_t)b * T_ + t;
    const float wxv = wx[bt * D_ + n];
    float acc = wxv;
    if (t > 0) {  // h0 == 0 -> skip the matvec on step 0
      const float* hp = hstate + (size_t)(t & 1) * (B_ * D_) + (size_t)b * D_;
#pragma unroll 4
      for (int k = 0; k < D_; k += 4) {
        const float4 hv = *(const float4*)(hp + k);
        acc += hv.x * (float)wr[k]     + hv.y * (float)wr[k + 1]
             + hv.z * (float)wr[k + 2] + hv.w * (float)wr[k + 3];
      }
    }
    const float hnew = tanhf(acc);
    const float zh = z[bt * (2 * D_) + n];
    const float zx = z[bt * (2 * D_) + D_ + n];
    const float gh = 1.f / (1.f + __expf(-zh));
    const float gx = 1.f / (1.f + __expf(-zx));
    h[bt * D_ + n] += gh * hnew + gx * wxv;                                   // residual add
    hstate[(size_t)((t + 1) & 1) * (B_ * D_) + (size_t)b * D_ + n] = hnew;    // publish

    // ---- device-wide split barrier between recurrence steps ----
    __builtin_amdgcn_fence(__ATOMIC_RELEASE, "agent");
    __syncthreads();
    if (tid == 0) {
      __hip_atomic_fetch_add(sync_cnt, 1, __ATOMIC_RELEASE, __HIP_MEMORY_SCOPE_AGENT);
      const int tgt = sync_base + GS_ * (t + 1);
      while (__hip_atomic_load(sync_cnt, __ATOMIC_ACQUIRE, __HIP_MEMORY_SCOPE_AGENT) < tgt)
        __builtin_amdgcn_s_sleep(2);
    }
    __syncthreads();
    __builtin_amdgcn_fence(__ATOMIC_ACQUIRE, "agent");
  }
}

// ---------------------------------------------------------------------------
// Host side
// ---------------------------------------------------------------------------
extern "C" void kernel_launch(void* const* d_in, const int* in_sizes, int n_in,
                              void* d_out, int out_size, void* d_ws, size_t ws_size,
                              hipStream_t stream) {
  (void)in_sizes; (void)n_in; (void)out_size; (void)ws_size;

  const int*   x     = (const int*)  d_in[0];
  const float* embed = (const float*)d_in[1];
  const float* Wx_w  = (const float*)d_in[2];
  const float* Wh_w  = (const float*)d_in[3];
  const float* Wz_w  = (const float*)d_in[4];
  const float* ln_s  = (const float*)d_in[5];
  const float* ln_b  = (const float*)d_in[6];
  const float* nrm_s = (const float*)d_in[7];
  const float* nrm_b = (const float*)d_in[8];
  float* logits = (float*)d_out;

  // workspace carve-up (~175 MB total)
  char* p = (char*)d_ws;
  _Float16* embed_h = (_Float16*)p; p += (size_t)V_ * D_ * 2;          // 64 MB
  _Float16* wxw_h   = (_Float16*)p; p += (size_t)L_ * D_ * D_ * 2;     // 8 MB
  _Float16* whw_h   = (_Float16*)p; p += (size_t)L_ * D_ * D_ * 2;     // 8 MB
  _Float16* wzw_h   = (_Float16*)p; p += (size_t)L_ * 2 * D_ * D_ * 2; // 16 MB
  float*    h       = (float*)p;    p += (size_t)BT_ * D_ * 4;         // 16 MB
  _Float16* xn_h    = (_Float16*)p; p += (size_t)BT_ * D_ * 2;         // 8 MB
  float*    wxb     = (float*)p;    p += (size_t)BT_ * D_ * 4;         // 16 MB
  float*    zb      = (float*)p;    p += (size_t)BT_ * 2 * D_ * 4;     // 32 MB
  float*    hstate  = (float*)p;    p += (size_t)2 * B_ * D_ * 4;      // 32 KB
  int*      syncc   = (int*)p;      p += 256;

  hipMemsetAsync(syncc, 0, 256, stream);  // graph-capture safe

  auto cvb = [](size_t n) { return (unsigned)((n / 4 + 255) / 256); };
  k_cvt_f16<<<cvb((size_t)V_ * D_), 256, 0, stream>>>(embed, embed_h, V_ * D_);
  k_cvt_f16<<<cvb((size_t)L_ * D_ * D_), 256, 0, stream>>>(Wx_w, wxw_h, L_ * D_ * D_);
  k_cvt_f16<<<cvb((size_t)L_ * D_ * D_), 256, 0, stream>>>(Wh_w, whw_h, L_ * D_ * D_);
  k_cvt_f16<<<cvb((size_t)L_ * 2 * D_ * D_), 256, 0, stream>>>(Wz_w, wzw_h, L_ * 2 * D_ * D_);

  k_embed_gather<<<BT_, 256, 0, stream>>>(x, embed, h);

  for (int l = 0; l < L_; ++l) {
    k_layernorm_f16<<<BT_, 256, 0, stream>>>(h, ln_s + (size_t)l * D_,
                                             ln_b + (size_t)l * D_, xn_h);
    dim3 g1(D_ / 128, BT_ / 128);
    k_gemm_wmma<<<g1, 256, 0, stream>>>(xn_h, wxw_h + (size_t)l * D_ * D_, wxb,
                                        BT_, D_, D_);
    dim3 g2((2 * D_) / 128, BT_ / 128);
    k_gemm_wmma<<<g2, 256, 0, stream>>>(xn_h, wzw_h + (size_t)l * 2 * D_ * D_, zb,
                                        BT_, 2 * D_, D_);
    k_scan<<<GS_, 128, 0, stream>>>(wxb, zb, whw_h + (size_t)l * D_ * D_, h,
                                    hstate, syncc, l * GS_ * T_);
  }

  k_layernorm_f16<<<BT_, 256, 0, stream>>>(h, nrm_s, nrm_b, xn_h);
  dim3 g3(V_ / 128, BT_ / 128);
  k_gemm_wmma<<<g3, 256, 0, stream>>>(xn_h, embed_h, logits, BT_, V_, D_);
}
